// TreeGRU_72559177499304
// MI455X (gfx1250) — matrix operand order repaired and checked by reading, not compiled, not run on hardware
//
#include <hip/hip_runtime.h>
#include <hip/hip_bf16.h>

typedef unsigned int       u32;
typedef unsigned short     u16;
typedef unsigned long long u64;

typedef __attribute__((ext_vector_type(16))) __bf16 v16bf;
typedef __attribute__((ext_vector_type(8)))  float  v8f;
typedef __attribute__((ext_vector_type(4)))  u32    v4u;
typedef __attribute__((ext_vector_type(8)))  u32    v8u;

union FragU { uint4 q[2]; v16bf v; };

#define NWG 16          // 16 workgroups * 32 h-cols = H=512
#define TT  512         // T
#define ROWP 136        // LDS row pitch in u16 (272B = 128 elems + 16B pad -> conflict-free)

// ---------- helpers ----------
__device__ __forceinline__ u16 f2bf(float f) {
    u32 u = __float_as_uint(f);
    u32 r = (u + 0x7FFFu + ((u >> 16) & 1u)) >> 16;   // RNE
    return (u16)r;
}
__device__ __forceinline__ float bf2f(u32 bits) {
    return __uint_as_float(bits << 16);
}
__device__ __forceinline__ float sigm(float x) {
    return 1.0f / (1.0f + __expf(-x));
}
__device__ __forceinline__ float tanh_fast(float x) {
    float e = __expf(-2.0f * x);
    return (1.0f - e) / (1.0f + e);
}

__device__ __forceinline__ v8f wmma_bf16(v16bf a, v16bf b, v8f c) {
    return __builtin_amdgcn_wmma_f32_16x16x32_bf16(false, a, false, b, (short)0, c, false, false);
}

// A fragment (16x32 bf16) from LDS (row pitch ROWP u16).
// 16-bit A layout: lane<16 -> row=lane,   K {0..7, 16..23}
//                  lane>=16 -> row=lane-16, K {8..15, 24..31}
__device__ __forceinline__ v16bf loadA(const u16* lds, int lane, int mtile, int ktl) {
    int r  = mtile * 16 + (lane & 15);
    int kb = ktl * 32 + ((lane & 16) ? 8 : 0);
    FragU f;
    f.q[0] = *(const uint4*)(lds + r * ROWP + kb);
    f.q[1] = *(const uint4*)(lds + r * ROWP + kb + 16);
    return f.v;
}

// B fragment (32x16 bf16) from pre-swizzled global weights:
// layout [nt][kt][lane][16 elems], 32B per lane.
__device__ __forceinline__ v16bf loadB(const u16* __restrict__ wf, int lane, int nt, int kt) {
    const u16* p = wf + ((size_t)(nt * 16 + kt) * 512) + lane * 16;
    FragU f;
    f.q[0] = *(const uint4*)p;
    f.q[1] = *(const uint4*)(p + 8);
    return f.v;
}

// ---------- setup: weights f32 -> bf16 WMMA-B fragment order; reset barrier counter ----------
__global__ __launch_bounds__(256) void convert_weights(const float* __restrict__ W_ih,
                                                       const float* __restrict__ W_hh,
                                                       u16* __restrict__ wfrag,
                                                       u32* __restrict__ cnt) {
    int e = blockIdx.x * 256 + threadIdx.x;            // 2 * 786432 elements
    if (e == 0) *cnt = 0u;
    int mat  = (e >= 786432);
    int r    = e - mat * 786432;
    int elem = r & 15;
    int lane = (r >> 4) & 31;
    int kt   = (r >> 9) & 15;
    int nt   = r >> 13;                                // 0..95
    int row  = nt * 16 + (lane & 15);                  // output gate column
    int kk   = kt * 32 + ((lane & 16) ? 8 : 0) + (elem & 7) + ((elem & 8) ? 16 : 0);
    const float* src = mat ? W_hh : W_ih;
    wfrag[e] = f2bf(src[(size_t)row * 512 + kk]);
}

// ---------- setup: xall[t][b][k] = bf16(embedding[child_ids[t,b]][k]) ----------
__global__ __launch_bounds__(256) void embed_gather(const float* __restrict__ emb,
                                                    const int* __restrict__ child,
                                                    u16* __restrict__ xall) {
    int t4  = blockIdx.x * 256 + threadIdx.x;          // 4,202,496 threads
    int idx = t4 * 4;
    int k   = idx & 511;
    int row = idx >> 9;                                // 0..32831  (t*64+b)
    int cid = child[row];
    float4 f = *(const float4*)(emb + (size_t)cid * 512 + k);
    uint2 o;
    o.x = (u32)f2bf(f.x) | ((u32)f2bf(f.y) << 16);
    o.y = (u32)f2bf(f.z) | ((u32)f2bf(f.w) << 16);
    *(uint2*)(xall + (size_t)row * 512 + k) = o;
}

// ---------- persistent recurrent kernel: 16 WGs, grid barrier per step ----------
__global__ __launch_bounds__(256) void treegru_recur(const float* __restrict__ b_ih,
                                                     const float* __restrict__ b_hh,
                                                     const int* __restrict__ parent,
                                                     const u16* __restrict__ xall,
                                                     const u16* __restrict__ wih_f,
                                                     const u16* __restrict__ whh_f,
                                                     u16* __restrict__ buf,
                                                     u32* __restrict__ cnt) {
    __shared__ u16 xs[64 * ROWP];   // 17,408 B  (x tile, written by TDM, padded rows)
    __shared__ u16 hs[64 * ROWP];   // 17,408 B  (h_prev tile, manual gather)

    const int tid  = threadIdx.x;
    const int w    = blockIdx.x;           // h-column chunk [32w, 32w+32)
    const int lane = tid & 31;
    const int wid  = tid >> 5;             // 8 waves
    const int mtile = wid >> 1;            // batch rows [16*mtile, +16)
    const int jsub  = wid & 1;             // which 16-col subtile
    const int ncol  = w * 32 + jsub * 16 + (lane & 15);   // this lane's C/D column (h index)
    const int rnt = 2 * w + jsub, znt = 32 + rnt, nnt = 64 + rnt;

    const float biasR  = b_ih[ncol]        + b_hh[ncol];
    const float biasZ  = b_ih[512 + ncol]  + b_hh[512 + ncol];
    const float biasIN = b_ih[1024 + ncol];
    const float biasHN = b_hh[1024 + ncol];

    const u32 lds_x = (u32)(u64)(&xs[0]);  // low 32 bits of flat shared addr = LDS offset

    for (int t = 0; t <= TT; ++t) {
        v8f ar, az, ain, ahn;
        for (int i = 0; i < 8; ++i) { ar[i] = biasR; az[i] = biasZ; ain[i] = biasIN; ahn[i] = biasHN; }

        for (int kq = 0; kq < 4; ++kq) {               // K staged in quarters of 128
            __syncthreads();                           // previous stage/ step LDS reads done

            if (wid == 0) {
                // TDM: 64 rows x 128 bf16, global row stride 512 elems, LDS rows padded
                // by 4 dwords every 64 dwords -> 272B pitch.
                u64 ga = (u64)(const void*)xall + ((u64)t * 32768ull + (u64)kq * 128ull) * 2ull;
                v4u g0;
                g0[0] = 1u;                                   // count=1, user descriptor
                g0[1] = lds_x;                                // lds_addr
                g0[2] = (u32)ga;                              // global_addr[31:0]
                g0[3] = ((u32)(ga >> 32) & 0x1FFFFFFu) | (2u << 30);  // addr[56:32] | type=2
                v8u g1;
                g1[0] = (1u << 16) | (1u << 20) | (5u << 22) | (3u << 25); // 2B, pad_en, 64dw interval, 4dw pad
                g1[1] = 0u;                                   // tensor_dim0[15:0]=0 (dim0=1<<30)
                g1[2] = 0x4000u;                              // tensor_dim0 hi
                g1[3] = 0x4000u | (128u << 16);               // tensor_dim1 hi | tile_dim0=128
                g1[4] = 64u;                                  // tile_dim1=64, tile_dim2=0
                g1[5] = 512u;                                 // tensor_dim0_stride lo (elems)
                g1[6] = 0u;
                g1[7] = 0u;
                asm volatile("tensor_load_to_lds %0, %1" :: "s"(g0), "s"(g1) : "memory");
            }

            { // h_prev gather: 64 rows x 128 bf16 from buf[parent_idx[t-1,b]]
                int row  = tid >> 2, part = tid & 3;
                int pt   = (t == 0) ? -1 : parent[(t - 1) * 64 + row];
                uint4* dst = (uint4*)(hs + row * ROWP + part * 32);
                if (pt >= 0) {
                    const uint4* src = (const uint4*)(buf + (size_t)pt * 32768 + row * 512 + kq * 128 + part * 32);
                    dst[0] = src[0]; dst[1] = src[1]; dst[2] = src[2]; dst[3] = src[3];
                } else {
                    uint4 z = make_uint4(0u, 0u, 0u, 0u);
                    dst[0] = z; dst[1] = z; dst[2] = z; dst[3] = z;
                }
            }

            if (wid == 0) __builtin_amdgcn_s_wait_tensorcnt(0);
            __syncthreads();

            for (int ktl = 0; ktl < 4; ++ktl) {
                int kt = kq * 4 + ktl;                 // absolute K tile 0..15
                v16bf aX = loadA(xs, lane, mtile, ktl);
                v16bf aH = loadA(hs, lane, mtile, ktl);
                ar  = wmma_bf16(aX, loadB(wih_f, lane, rnt, kt), ar);
                ar  = wmma_bf16(aH, loadB(whh_f, lane, rnt, kt), ar);
                az  = wmma_bf16(aX, loadB(wih_f, lane, znt, kt), az);
                az  = wmma_bf16(aH, loadB(whh_f, lane, znt, kt), az);
                ain = wmma_bf16(aX, loadB(wih_f, lane, nnt, kt), ain);
                ahn = wmma_bf16(aH, loadB(whh_f, lane, nnt, kt), ahn);
            }
        }

        // epilogue: C/D layout lane -> N=lane&15, M = v + (lane&16 ? 8 : 0)
        for (int v = 0; v < 8; ++v) {
            int m  = mtile * 16 + v + ((lane & 16) ? 8 : 0);
            int pt = (t == 0) ? -1 : parent[(t - 1) * 64 + m];
            float hp = 0.0f;
            if (pt >= 0) hp = bf2f((u32)buf[(size_t)pt * 32768 + m * 512 + ncol]);
            float r = sigm(ar[v]);
            float z = sigm(az[v]);
            float n = tanh_fast(ain[v] + r * ahn[v]);
            float h = (1.0f - z) * n + z * hp;
            buf[(size_t)t * 32768 + m * 512 + ncol] = f2bf(h);
        }

        // grid-wide barrier between steps
        __builtin_amdgcn_fence(__ATOMIC_RELEASE, "agent");
        __syncthreads();
        if (tid == 0) {
            atomicAdd(cnt, 1u);
            u32 target = (u32)(NWG * (t + 1));
            while (__hip_atomic_load(cnt, __ATOMIC_RELAXED, __HIP_MEMORY_SCOPE_AGENT) < target)
                __builtin_amdgcn_s_sleep(1);
        }
        __syncthreads();
        __builtin_amdgcn_fence(__ATOMIC_ACQUIRE, "agent");
    }
}

// ---------- leaf gather: out[b,l,:] = buf[leaf_idx[b,l], b, :] (f32) ----------
__global__ __launch_bounds__(256) void leaf_gather(const int* __restrict__ leaf,
                                                   const u16* __restrict__ buf,
                                                   float* __restrict__ out) {
    int t4  = blockIdx.x * 256 + threadIdx.x;          // 2,097,152 threads
    int idx = t4 * 4;
    int k = idx & 511;
    int l = (idx >> 9) & 255;
    int b = idx >> 17;
    int li = leaf[b * 256 + l];
    float4 o;
    if (li < 0) {
        o = make_float4(0.f, 0.f, 0.f, 0.f);
    } else {
        uint2 q = *(const uint2*)(buf + (size_t)li * 32768 + b * 512 + k);
        o.x = bf2f(q.x & 0xFFFFu); o.y = bf2f(q.x >> 16);
        o.z = bf2f(q.y & 0xFFFFu); o.w = bf2f(q.y >> 16);
    }
    *(float4*)(out + (size_t)idx) = o;
}

extern "C" void kernel_launch(void* const* d_in, const int* in_sizes, int n_in,
                              void* d_out, int out_size, void* d_ws, size_t ws_size,
                              hipStream_t stream) {
    const float* emb    = (const float*)d_in[0];
    const float* W_ih   = (const float*)d_in[1];
    const float* W_hh   = (const float*)d_in[2];
    const float* b_ih   = (const float*)d_in[3];
    const float* b_hh   = (const float*)d_in[4];
    const int*   child  = (const int*)d_in[5];
    const int*   parent = (const int*)d_in[6];
    const int*   leaf   = (const int*)d_in[7];
    float* out = (float*)d_out;

    char* ws = (char*)d_ws;
    u32* cnt   = (u32*)ws;                                   // barrier counter
    u16* wfrag = (u16*)(ws + 1024);                          // 3,145,728 B (both matrices)
    u16* xall  = (u16*)(ws + 1024 + 3145728);                // 33,619,968 B
    u16* buf   = (u16*)(ws + 1024 + 3145728 + 33619968);     // 33,619,968 B

    convert_weights<<<dim3(6144), dim3(256), 0, stream>>>(W_ih, W_hh, wfrag, cnt);
    embed_gather   <<<dim3(16416), dim3(256), 0, stream>>>(emb, child, xall);
    treegru_recur  <<<dim3(NWG),  dim3(256), 0, stream>>>(b_ih, b_hh, parent, xall,
                                                          wfrag, wfrag + 786432, buf, cnt);
    leaf_gather    <<<dim3(8192), dim3(256), 0, stream>>>(leaf, buf, out);
}
